// RoutedLoRALinear_9680856285464
// MI455X (gfx1250) — matrix operand order repaired and checked by reading, not compiled, not run on hardware
//
#include <hip/hip_runtime.h>
#include <hip/hip_bf16.h>
#include <stdint.h>

// ---------------------------------------------------------------------------
// RoutedLoRALinear for MI455X (gfx1250, wave32, WMMA)
//   y[m,n] = sum_k x[m,k]*W[n,k] + bias[n]
//          + SCALING * sum_r ( sum_k x[m,k]*A[e,r,k] ) * Bp[e,n,r],  e=role[m]
// Shapes (hardcoded from setup_inputs): M=8*4096=32768, K=D=1024, N=O=1024,
// E=8, R=16, SCALING=2.0f.
// ---------------------------------------------------------------------------

typedef __attribute__((ext_vector_type(16))) __bf16 v16bf;
typedef __attribute__((ext_vector_type(8)))  float  v8f;

#define M_TOK   32768
#define KDIM    1024
#define NDIM    1024
#define RANK    16
#define SCALING 2.0f

#define BM 128
#define BN 128
#define BK 32
#define LDST 17   // padded LDS row stride in uints (16 K-pairs + 1 pad)

// round-to-nearest-even f32 -> bf16, packed pair (lo = a, hi = b)
__device__ inline uint32_t pack_bf16x2(float a, float b) {
  union { float f; uint32_t u; } ua, ub;
  ua.f = a; ub.f = b;
  uint32_t ra = ua.u + 0x7FFFu + ((ua.u >> 16) & 1u);
  uint32_t rb = ub.u + 0x7FFFu + ((ub.u >> 16) & 1u);
  return (ra >> 16) | (rb & 0xFFFF0000u);
}

// ---------------------------------------------------------------------------
// Kernel 1: per-token low-rank down-projection h[m,r] = sum_k x[m,k]*A[e,r,k]
// One wave (32 lanes) per token; 8 waves per 256-thread block.
// ---------------------------------------------------------------------------
__global__ __launch_bounds__(256)
void lora_down_kernel(const float* __restrict__ x,
                      const int*   __restrict__ role,
                      const float* __restrict__ A,
                      float*       __restrict__ h) {
  const int token = blockIdx.x * 8 + (threadIdx.x >> 5);
  const int lane  = threadIdx.x & 31;
  const int e     = role[token];

  const float* __restrict__ xrow = x + (size_t)token * KDIM;
  const float* __restrict__ Ae   = A + (size_t)e * RANK * KDIM;

  float acc[RANK];
#pragma unroll
  for (int r = 0; r < RANK; ++r) acc[r] = 0.0f;

#pragma unroll
  for (int it = 0; it < KDIM / 128; ++it) {
    const int d = it * 128 + lane * 4;
    const float4 xv = *(const float4*)(xrow + d);
#pragma unroll
    for (int r = 0; r < RANK; ++r) {
      const float4 av = *(const float4*)(Ae + r * KDIM + d);
      acc[r] += xv.x * av.x + xv.y * av.y + xv.z * av.z + xv.w * av.w;
    }
  }

  // cross-lane reduction within the wave32
#pragma unroll
  for (int r = 0; r < RANK; ++r) {
    float v = acc[r];
#pragma unroll
    for (int off = 16; off > 0; off >>= 1) v += __shfl_xor(v, off, 32);
    acc[r] = v;
  }
  if (lane < RANK) h[(size_t)token * RANK + lane] = acc[lane];
}

// ---------------------------------------------------------------------------
// Kernel 2: fused GEMM (bf16 WMMA, f32 accumulate) + bias + routed LoRA epilogue
// Block: 256 threads (8 waves). Block tile 128x128, K-step 32.
// Wave grid 2(M) x 4(N): each wave owns 64x32 = 4x2 tiles of 16x16.
// ---------------------------------------------------------------------------
__global__ __launch_bounds__(256)
void fused_gemm_kernel(const float* __restrict__ x,
                       const int*   __restrict__ role,
                       const float* __restrict__ W,
                       const float* __restrict__ bias,
                       const float* __restrict__ Bp,
                       const float* __restrict__ h,
                       float*       __restrict__ out) {
  __shared__ uint32_t Xs[BM * LDST];   // 128 rows x 16 K-pairs (bf16x2) + pad
  __shared__ uint32_t Ws[BN * LDST];

  const int tid  = threadIdx.x;
  const int wave = tid >> 5;
  const int lane = tid & 31;

  const int row_base = blockIdx.x * BM;   // 0..32768 step 128
  const int col_base = blockIdx.y * BN;   // 0..1024  step 128

  const int wrow = (wave >> 2) * 64;      // wave M offset within block tile
  const int wcol = (wave & 3) * 32;       // wave N offset within block tile

  v8f zero;
#pragma unroll
  for (int t = 0; t < 8; ++t) zero[t] = 0.0f;

  v8f acc[4][2];
#pragma unroll
  for (int i = 0; i < 4; ++i)
#pragma unroll
    for (int j = 0; j < 2; ++j) acc[i][j] = zero;

  const int ldr = tid >> 3;   // loader row 0..31 (+32 per pass)
  const int ldc = tid & 7;    // loader float4 column 0..7

  for (int kt = 0; kt < KDIM; kt += BK) {
    __syncthreads();
    // ---- stage X tile (f32 -> packed bf16 pairs) ----
#pragma unroll
    for (int p = 0; p < 4; ++p) {
      const int r = ldr + p * 32;
      const float4 v = *(const float4*)(x + (size_t)(row_base + r) * KDIM + kt + ldc * 4);
      Xs[r * LDST + ldc * 2]     = pack_bf16x2(v.x, v.y);
      Xs[r * LDST + ldc * 2 + 1] = pack_bf16x2(v.z, v.w);
    }
    // ---- stage W tile ----
#pragma unroll
    for (int p = 0; p < 4; ++p) {
      const int r = ldr + p * 32;
      const float4 v = *(const float4*)(W + (size_t)(col_base + r) * KDIM + kt + ldc * 4);
      Ws[r * LDST + ldc * 2]     = pack_bf16x2(v.x, v.y);
      Ws[r * LDST + ldc * 2 + 1] = pack_bf16x2(v.z, v.w);
    }
    // prefetch next K-tile into caches (global_prefetch_b8)
    if (kt + BK < KDIM) {
      __builtin_prefetch(x + (size_t)(row_base + ldr) * KDIM + kt + BK + ldc * 4, 0, 1);
      __builtin_prefetch(W + (size_t)(col_base + ldr) * KDIM + kt + BK + ldc * 4, 0, 1);
    }
    __syncthreads();

    // ---- build fragments from LDS per ISA VGPR layouts ----
    const int mlo = lane & 15;
    const int hlf = lane >> 4;

    union Frag { uint32_t u[8]; v16bf v; };
    Frag a[4], b[2];
#pragma unroll
    for (int i = 0; i < 4; ++i) {
      const int m = wrow + i * 16 + mlo;
#pragma unroll
      for (int j = 0; j < 8; ++j) {
        const int grp = j >> 2;
        const int kp  = grp * 8 + hlf * 4 + (j & 3);   // A 16x32 bf16 layout
        a[i].u[j] = Xs[m * LDST + kp];
      }
    }
#pragma unroll
    for (int i = 0; i < 2; ++i) {
      const int n = wcol + i * 16 + mlo;
#pragma unroll
      for (int j = 0; j < 8; ++j) {
        const int kp = (j >> 2) * 8 + hlf * 4 + (j & 3); // B 32x16 bf16 layout
        b[i].u[j] = Ws[n * LDST + kp];
      }
    }

    // ---- 8 WMMAs: 64x32 output per wave per K-step ----
#pragma unroll
    for (int i = 0; i < 4; ++i)
#pragma unroll
      for (int jn = 0; jn < 2; ++jn)
        acc[i][jn] = __builtin_amdgcn_wmma_f32_16x16x32_bf16(
            false, a[i].v, false, b[jn].v, (short)0, acc[i][jn], false, false);
  }

  // ---- epilogue: bias + routed LoRA up-projection, f32 store ----
  const int mlo = lane & 15;
  const int hlf = lane >> 4;
#pragma unroll
  for (int i = 0; i < 4; ++i) {
#pragma unroll
    for (int jn = 0; jn < 2; ++jn) {
      const int n  = col_base + wcol + jn * 16 + mlo;
      const float bv = bias[n];
#pragma unroll
      for (int j = 0; j < 8; ++j) {
        const int m = row_base + wrow + i * 16 + hlf * 8 + j;
        const int e = role[m];
        const float* __restrict__ hr = h  + (size_t)m * RANK;
        const float* __restrict__ bp = Bp + ((size_t)e * NDIM + n) * RANK;
        const float4 h0 = *(const float4*)(hr);
        const float4 h1 = *(const float4*)(hr + 4);
        const float4 h2 = *(const float4*)(hr + 8);
        const float4 h3 = *(const float4*)(hr + 12);
        const float4 b0 = *(const float4*)(bp);
        const float4 b1 = *(const float4*)(bp + 4);
        const float4 b2 = *(const float4*)(bp + 8);
        const float4 b3 = *(const float4*)(bp + 12);
        float lora = h0.x * b0.x + h0.y * b0.y + h0.z * b0.z + h0.w * b0.w
                   + h1.x * b1.x + h1.y * b1.y + h1.z * b1.z + h1.w * b1.w
                   + h2.x * b2.x + h2.y * b2.y + h2.z * b2.z + h2.w * b2.w
                   + h3.x * b3.x + h3.y * b3.y + h3.z * b3.z + h3.w * b3.w;
        out[(size_t)m * NDIM + n] = acc[i][jn][j] + bv + SCALING * lora;
      }
    }
  }
}

// ---------------------------------------------------------------------------
extern "C" void kernel_launch(void* const* d_in, const int* in_sizes, int n_in,
                              void* d_out, int out_size, void* d_ws, size_t ws_size,
                              hipStream_t stream) {
  (void)in_sizes; (void)n_in; (void)out_size; (void)ws_size;
  const float* x    = (const float*)d_in[0];
  const int*   role = (const int*)  d_in[1];
  const float* W    = (const float*)d_in[2];
  const float* bias = (const float*)d_in[3];
  const float* A    = (const float*)d_in[4];
  const float* Bp   = (const float*)d_in[5];
  float* out = (float*)d_out;
  float* h   = (float*)d_ws;   // 32768 * 16 * 4 = 2 MB scratch

  lora_down_kernel<<<M_TOK / 8, 256, 0, stream>>>(x, role, A, h);

  dim3 grid(M_TOK / BM, NDIM / BN);   // (256, 8)
  fused_gemm_kernel<<<grid, 256, 0, stream>>>(x, role, W, bias, Bp, h, out);
}
